// SessionGraph_67551245632223
// MI455X (gfx1250) — matrix operand order repaired and checked by reading, not compiled, not run on hardware
//
#include <hip/hip_runtime.h>
#include <hip/hip_bf16.h>

// ---------------- problem constants ----------------
#define B_   128
#define L_   50
#define D_   128
#define S_   12           // SAMPLE_NUM
#define M1_  (L_ * S_)    // 600
#define M2_  (M1_ * S_)   // 7200
#define LP_  64           // padded L for WMMA tiles
#define NREL 25           // 2*MAX_REL+1
#define ALPHA_ 0.2f

typedef __attribute__((ext_vector_type(16))) _Float16 v16h;
typedef __attribute__((ext_vector_type(8)))  _Float16 v8h;
typedef __attribute__((ext_vector_type(4)))  _Float16 v4h;
typedef __attribute__((ext_vector_type(8)))  float    v8f;
typedef __attribute__((ext_vector_type(4)))  unsigned int v4u;
typedef __attribute__((ext_vector_type(8)))  int v8i;
typedef __attribute__((ext_vector_type(4)))  int v4i;

#define DEV __device__ __forceinline__

// D = A(16x32 f16) x B(32x16 f16) + C (f32), wave32
DEV v8f wmma32(v16h a, v16h b, v8f c) {
  return __builtin_amdgcn_wmma_f32_16x16x32_f16(
      /*neg_a=*/false, a, /*neg_b=*/false, b,
      /*c_mod=*/(short)0, c, /*reuse_a=*/false, /*reuse_b=*/false);
}

// A-fragment from a row-major f16 row pointer.
// Lane (m = lane&15) owns row m; half = lane>>4 selects K-chunks
// [k0+half*16, +8) and [k0+half*16+8, +8): two contiguous 16B loads.
DEV v16h load_a_frag(const _Float16* row, int k0, int half) {
  v8h lo = *(const v8h*)(row + k0 + half * 16);
  v8h hi = *(const v8h*)(row + k0 + half * 16 + 8);
  v16h f;
#pragma unroll
  for (int i = 0; i < 8; ++i) { f[i] = lo[i]; f[8 + i] = hi[i]; }
  return f;
}

// B-fragment for C = X @ Y^T where Y is row-major f16: element e has
// K = k0 + half*16 + e, N = n0 + (lane&15); B[K][N] = Y[N][K] -> contiguous.
DEV v16h load_bT_frag(const _Float16* Yrow_n, int k0, int half) {
  return *(const v16h*)(Yrow_n + k0 + half * 16);
}

DEV v16h vmul16(v16h a, v16h b) {
#pragma unroll
  for (int i = 0; i < 16; ++i) a[i] = a[i] * b[i];
  return a;
}

DEV v8f vzero8() { v8f c = {0.f,0.f,0.f,0.f,0.f,0.f,0.f,0.f}; return c; }

// ---- Tensor Data Mover: 1-D contiguous copy global -> LDS (8B elements) ----
// D# per CDNA5 ISA ch.8: group0 = {count/type/addr}, group1 = {dims/strides}.
// nbytes must be a multiple of 8. One wave issues this; completion tracked
// by TENSORcnt (s_wait_tensorcnt 0).
DEV void tdm_load_to_lds(unsigned lds_off, const void* gptr, unsigned nbytes) {
  unsigned long long ga = (unsigned long long)gptr;
  unsigned n8 = nbytes >> 3;                    // element count (data_size = 8B)
  v4u g0;
  g0[0] = 1u;                                   // count=1, user descriptor
  g0[1] = lds_off;                              // lds_addr (bytes)
  g0[2] = (unsigned)(ga & 0xffffffffu);         // global_addr[31:0]
  g0[3] = (unsigned)((ga >> 32) & 0x1ffffffu)   // global_addr[56:32]
        | (2u << 30);                           // type = 2 ("image")
  v8i g1;
  g1[0] = (int)(3u << 16);                      // data_size=3 (8B), wg_mask=0
  g1[1] = (int)((n8 & 0xffffu) << 16);          // tensor_dim0[15:0] @ bits 63:48
  g1[2] = (int)(((n8 >> 16) & 0xffffu))         // tensor_dim0[31:16]
        | (1 << 16);                            // tensor_dim1 = 1 (low 16)
  g1[3] = (int)((n8 & 0xffffu) << 16);          // tile_dim0 = n8 @ bits 127:112
  g1[4] = 1;                                    // tile_dim1 = 1, tile_dim2 = 0
  g1[5] = (int)n8;                              // tensor_dim0_stride[31:0]
  g1[6] = 0;                                    // stride hi / dim1_stride lo
  g1[7] = 0;
  v4i z4 = {0, 0, 0, 0};
#if __clang_major__ >= 23
  v8i z8 = {0, 0, 0, 0, 0, 0, 0, 0};
  __builtin_amdgcn_tensor_load_to_lds(g0, g1, z4, z4, z8, 0);
#else
  __builtin_amdgcn_tensor_load_to_lds(g0, g1, z4, z4, 0);
#endif
}

DEV unsigned lds_offset_of(const void* p) {
  // generic pointer to LDS: addr[31:0] is the LDS byte offset
  return (unsigned)(unsigned long long)p;
}

// ---------------- gather / convert ----------------
__global__ void k_prep(const int* __restrict__ inputs, const int* __restrict__ item,
                       const int* __restrict__ mask, const float* __restrict__ emb,
                       _Float16* __restrict__ h16, _Float16* __restrict__ it16,
                       float* __restrict__ it32) {
  int i = blockIdx.x, d = threadIdx.x;
  int nh = inputs[i], ni = item[i];
  float mk = (float)mask[i];
  float hv = emb[(size_t)nh * D_ + d];
  float iv = emb[(size_t)ni * D_ + d] * mk;
  h16[(size_t)i * D_ + d]  = (_Float16)hv;
  it16[(size_t)i * D_ + d] = (_Float16)iv;
  it32[(size_t)i * D_ + d] = iv;
}

__global__ void k_mean(const float* __restrict__ it32, const int* __restrict__ mask,
                       float* __restrict__ s_b) {
  int b = blockIdx.x, d = threadIdx.x;
  float s = 0.f, c = 0.f;
  for (int l = 0; l < L_; ++l) {
    s += it32[((size_t)b * L_ + l) * D_ + d];
    c += (float)mask[b * L_ + l];
  }
  s_b[(size_t)b * D_ + d] = s / c;
}

// 4 rows per block; one float4 -> 4xf16 per lane
__global__ void k_gather1(const int* __restrict__ first_adj, const float* __restrict__ emb,
                          _Float16* __restrict__ ev1) {
  int t = threadIdx.x;
  int r = blockIdx.x * 4 + (t >> 5);
  int c4 = (t & 31) * 4;
  int node = first_adj[r];
  float4 f = *(const float4*)(emb + (size_t)node * D_ + c4);
  v4h h; h[0] = (_Float16)f.x; h[1] = (_Float16)f.y; h[2] = (_Float16)f.z; h[3] = (_Float16)f.w;
  *(v4h*)(ev1 + (size_t)r * D_ + c4) = h;
}

__global__ void k_gather2(const int* __restrict__ first_adj, const int* __restrict__ adj_all,
                          const float* __restrict__ emb, _Float16* __restrict__ ev2) {
  int t = threadIdx.x;
  int r = blockIdx.x * 4 + (t >> 5);
  int c4 = (t & 31) * 4;
  int b = r / M2_, m2 = r % M2_;
  int m = m2 / S_, j = m2 % S_;
  int n1 = first_adj[b * M1_ + m];
  int node = adj_all[(size_t)n1 * S_ + j];
  float4 f = *(const float4*)(emb + (size_t)node * D_ + c4);
  v4h h; h[0] = (_Float16)f.x; h[1] = (_Float16)f.y; h[2] = (_Float16)f.z; h[3] = (_Float16)f.w;
  *(v4h*)(ev2 + (size_t)r * D_ + c4) = h;
}

// ---------------- weight pre-pack (B-fragment layout) ----------------
// packed[((kt*NT + nt)*32 + lane)*16 + e] = W[kt*32 + (lane>>4)*16 + e][nt*16 + (lane&15)]
__global__ void k_packw1(const float* __restrict__ w1, const float* __restrict__ s_b,
                         _Float16* __restrict__ pb) {
  int hop = blockIdx.x >> 7, b = blockIdx.x & 127;
  const float* W = w1 + (size_t)hop * D_ * D_;
  const float* s = s_b + (size_t)b * D_;
  _Float16* out = pb + (size_t)blockIdx.x * (4 * 8 * 32 * 16);
  for (int idx = threadIdx.x; idx < 4 * 8 * 32 * 16; idx += blockDim.x) {
    int e = idx & 15, lane = (idx >> 4) & 31, nt = (idx >> 9) & 7, kt = idx >> 12;
    int K = kt * 32 + (lane >> 4) * 16 + e;
    int N = nt * 16 + (lane & 15);
    out[idx] = (_Float16)(s[K] * W[K * D_ + N]);   // fold diag(s_b) into w1
  }
}

__global__ void k_packw3(const float* __restrict__ w3, _Float16* __restrict__ pb) {
  int hop = blockIdx.x;
  const float* W = w3 + (size_t)hop * 2 * D_ * D_;   // 256 x 128
  _Float16* out = pb + (size_t)hop * (8 * 8 * 32 * 16);
  for (int idx = threadIdx.x; idx < 8 * 8 * 32 * 16; idx += blockDim.x) {
    int e = idx & 15, lane = (idx >> 4) & 31, nt = (idx >> 9) & 7, kt = idx >> 12;
    int K = kt * 32 + (lane >> 4) * 16 + e;
    int N = nt * 16 + (lane & 15);
    out[idx] = (_Float16)W[K * D_ + N];
  }
}

// ---------------- LocalAggregator (per batch, 8 waves) ----------------
__global__ void k_local(const _Float16* __restrict__ h16, const int* __restrict__ adj,
                        const float* __restrict__ local_a, float* __restrict__ out) {
  __shared__ __align__(32) _Float16 hs[LP_][D_];   // 16 KB
  __shared__ float es[LP_][LP_];                   // 16 KB
  __shared__ float alf[LP_][LP_];                  // 16 KB
  __shared__ __align__(32) _Float16 als[LP_][LP_]; //  8 KB
  int b = blockIdx.x, tid = threadIdx.x;
  int lane = tid & 31, wave = tid >> 5;
  int m = lane & 15, half = lane >> 4;

  for (int idx = tid; idx < LP_ * D_; idx += 256) {
    int l = idx >> 7, d = idx & 127;
    hs[l][d] = (l < L_) ? h16[((size_t)b * L_ + l) * D_ + d] : (_Float16)0.f;
  }
  for (int idx = tid; idx < LP_ * LP_; idx += 256) {
    alf[idx >> 6][idx & 63] = -9e15f;
    als[idx >> 6][idx & 63] = (_Float16)0.f;
  }
  __syncthreads();

  // e_z = (h .* a_z) @ h^T, z = 0..3; select into alf by edge type
  for (int z = 0; z < 4; ++z) {
    v16h laf[4];
#pragma unroll
    for (int kt = 0; kt < 4; ++kt) {
      v16h f;
#pragma unroll
      for (int i = 0; i < 8; ++i) {
        f[i]     = (_Float16)local_a[(kt * 32 + half * 16 + i) * 4 + z];
        f[8 + i] = (_Float16)local_a[(kt * 32 + half * 16 + 8 + i) * 4 + z];
      }
      laf[kt] = f;
    }
    for (int t = wave; t < 16; t += 8) {
      int mt = t >> 2, nt = t & 3;
      v8f c = vzero8();
#pragma unroll
      for (int kt = 0; kt < 4; ++kt) {
        v16h a = vmul16(load_a_frag(&hs[mt * 16 + m][0], kt * 32, half), laf[kt]);
        v16h bf = load_bT_frag(&hs[nt * 16 + m][0], kt * 32, half);
        c = wmma32(a, bf, c);
      }
#pragma unroll
      for (int i = 0; i < 8; ++i) es[mt * 16 + half * 8 + i][nt * 16 + m] = c[i];
    }
    __syncthreads();
    for (int idx = tid; idx < L_ * L_; idx += 256) {
      int l = idx / L_, k = idx % L_;
      if (adj[((size_t)b * L_ + l) * L_ + k] == z + 1) {
        float v = es[l][k];
        alf[l][k] = v > 0.f ? v : ALPHA_ * v;      // leaky_relu on e
      }
    }
    __syncthreads();
  }

  if (tid < L_) {
    float mx = -3.0e38f;
    for (int k = 0; k < L_; ++k) mx = fmaxf(mx, alf[tid][k]);
    float s = 0.f;
    for (int k = 0; k < L_; ++k) s += __expf(alf[tid][k] - mx);
    float inv = 1.f / s;
    for (int k = 0; k < L_; ++k) als[tid][k] = (_Float16)(__expf(alf[tid][k] - mx) * inv);
  }
  __syncthreads();

  // h_local = alpha(64x64) @ h(64x128)
  for (int t = wave; t < 32; t += 8) {
    int mt = t & 3, nt = t >> 2;
    v8f c = vzero8();
#pragma unroll
    for (int kt = 0; kt < 2; ++kt) {
      v16h a = load_a_frag(&als[mt * 16 + m][0], kt * 32, half);
      v16h bf;
#pragma unroll
      for (int e = 0; e < 16; ++e) bf[e] = hs[kt * 32 + half * 16 + e][nt * 16 + m];
      c = wmma32(a, bf, c);
    }
#pragma unroll
    for (int i = 0; i < 8; ++i) {
      int l = mt * 16 + half * 8 + i, d = nt * 16 + m;
      if (l < L_) out[((size_t)b * L_ + l) * D_ + d] = c[i];
    }
  }
}

// ---------------- rel-pos self attention (per batch, 8 waves) ----------------
__global__ void k_attn(const _Float16* __restrict__ it16, const float* __restrict__ rel,
                       float* __restrict__ out2) {
  __shared__ __align__(32) _Float16 its[LP_][D_];   // 16 KB
  __shared__ __align__(32) _Float16 rels[32][D_];   //  8 KB
  __shared__ float at[LP_][LP_];                    // 16 KB
  __shared__ float Ps[LP_][32];                     //  8 KB
  __shared__ __align__(32) _Float16 als[LP_][LP_];  //  8 KB
  int b = blockIdx.x, tid = threadIdx.x;
  int lane = tid & 31, wave = tid >> 5;
  int m = lane & 15, half = lane >> 4;

  for (int idx = tid; idx < LP_ * D_; idx += 256) {
    int l = idx >> 7, d = idx & 127;
    its[l][d] = (l < L_) ? it16[((size_t)b * L_ + l) * D_ + d] : (_Float16)0.f;
  }
  for (int idx = tid; idx < 32 * D_; idx += 256) {
    int r = idx >> 7, d = idx & 127;
    rels[r][d] = (r < NREL) ? (_Float16)rel[r * D_ + d] : (_Float16)0.f;
  }
  for (int idx = tid; idx < LP_ * LP_; idx += 256) als[idx >> 6][idx & 63] = (_Float16)0.f;
  __syncthreads();

  // attn1 = item @ item^T
  for (int t = wave; t < 16; t += 8) {
    int mt = t >> 2, nt = t & 3;
    v8f c = vzero8();
#pragma unroll
    for (int kt = 0; kt < 4; ++kt) {
      v16h a = load_a_frag(&its[mt * 16 + m][0], kt * 32, half);
      v16h bf = load_bT_frag(&its[nt * 16 + m][0], kt * 32, half);
      c = wmma32(a, bf, c);
    }
#pragma unroll
    for (int i = 0; i < 8; ++i) at[mt * 16 + half * 8 + i][nt * 16 + m] = c[i];
  }
  // P = item @ rel^T  (25 distinct rel vectors, padded to 32)
  {
    int mt = wave >> 1, nt = wave & 1;
    v8f c = vzero8();
#pragma unroll
    for (int kt = 0; kt < 4; ++kt) {
      v16h a = load_a_frag(&its[mt * 16 + m][0], kt * 32, half);
      v16h bf = load_bT_frag(&rels[nt * 16 + m][0], kt * 32, half);
      c = wmma32(a, bf, c);
    }
#pragma unroll
    for (int i = 0; i < 8; ++i) Ps[mt * 16 + half * 8 + i][nt * 16 + m] = c[i];
  }
  __syncthreads();

  const float isd = 0.08838834764831845f;   // 1/sqrt(128)
  if (tid < L_) {
    int q = tid;
    float mx = -3.0e38f;
    for (int k = 0; k < L_; ++k) {
      int r = k - q; r = r < -12 ? -12 : (r > 12 ? 12 : r); r += 12;
      float v = (at[q][k] + Ps[q][r]) * isd;
      at[q][k] = v;
      mx = fmaxf(mx, v);
    }
    float s = 0.f;
    for (int k = 0; k < L_; ++k) s += __expf(at[q][k] - mx);
    float inv = 1.f / s;
    for (int k = 0; k < L_; ++k) als[q][k] = (_Float16)(__expf(at[q][k] - mx) * inv);
  }
  __syncthreads();

  for (int t = wave; t < 32; t += 8) {
    int mt = t & 3, nt = t >> 2;
    v8f c = vzero8();
#pragma unroll
    for (int kt = 0; kt < 2; ++kt) {
      v16h a = load_a_frag(&als[mt * 16 + m][0], kt * 32, half);
      v16h bf;
#pragma unroll
      for (int e = 0; e < 16; ++e) bf[e] = its[kt * 32 + half * 16 + e][nt * 16 + m];
      c = wmma32(a, bf, c);
    }
#pragma unroll
    for (int i = 0; i < 8; ++i) {
      int l = mt * 16 + half * 8 + i, d = nt * 16 + m;
      if (l < L_) out2[((size_t)b * L_ + l) * D_ + d] = c[i];
    }
  }
}

// ---------------- GlobalAggregator score GEMM ----------------
// 8 waves/block, 8 row-tiles of one batch; packed W1_b staged to LDS by TDM.
// scores[r] = leakyrelu(neigh[r] @ diag(s_b) @ w1) @ w2
__global__ void k_score(const _Float16* __restrict__ A, const _Float16* __restrict__ pbW1,
                        const float* __restrict__ w2, float* __restrict__ scores,
                        int rowsPerBatch, int tilesPerBatch, int blocksPerBatch) {
  __shared__ __align__(32) _Float16 wlds[16384];   // 32 KB packed W1_b
  int tid = threadIdx.x;
  int lane = tid & 31, wave = tid >> 5;
  int b = blockIdx.x / blocksPerBatch;
  int tblk = blockIdx.x % blocksPerBatch;
  if (wave == 0) {
    tdm_load_to_lds(lds_offset_of(wlds), pbW1 + (size_t)b * 16384, 32768u);
    __builtin_amdgcn_s_wait_tensorcnt(0);
  }
  __syncthreads();
  int tile = tblk * 8 + wave;
  if (tile >= tilesPerBatch) return;
  int m = lane & 15, half = lane >> 4;
  int rloc = tile * 16 + m;
  if (rloc > rowsPerBatch - 1) rloc = rowsPerBatch - 1;   // clamp tail rows
  const _Float16* Arow = A + ((size_t)b * rowsPerBatch + rloc) * D_;
  v16h afr[4];
#pragma unroll
  for (int kt = 0; kt < 4; ++kt) afr[kt] = load_a_frag(Arow, kt * 32, half);
  float rsum[8];
#pragma unroll
  for (int i = 0; i < 8; ++i) rsum[i] = 0.f;
#pragma unroll
  for (int nt = 0; nt < 8; ++nt) {
    v8f c = vzero8();
#pragma unroll
    for (int kt = 0; kt < 4; ++kt) {
      v16h bf = *(const v16h*)(wlds + (((kt * 8 + nt) * 32 + lane) * 16));
      c = wmma32(afr[kt], bf, c);
    }
    float w2c = w2[nt * 16 + m];   // this lane's column weight
#pragma unroll
    for (int i = 0; i < 8; ++i) {
      float v = c[i];
      v = v > 0.f ? v : ALPHA_ * v;   // leaky_relu
      rsum[i] += v * w2c;
    }
  }
  // butterfly-reduce across the 16 lanes of each half (wave32 shuffles)
#pragma unroll
  for (int i = 0; i < 8; ++i) {
    float v = rsum[i];
    v += __shfl_xor(v, 8, 32);
    v += __shfl_xor(v, 4, 32);
    v += __shfl_xor(v, 2, 32);
    v += __shfl_xor(v, 1, 32);
    if ((lane & 15) == 0) {
      int rl = tile * 16 + half * 8 + i;
      if (rl < rowsPerBatch) scores[(size_t)b * rowsPerBatch + rl] = v;
    }
  }
}

// softmax over 12 neighbors + weighted aggregate
__global__ void k_softagg(const float* __restrict__ scores, const _Float16* __restrict__ neigh,
                          _Float16* __restrict__ agg) {
  __shared__ float al[S_];
  int g = blockIdx.x, d = threadIdx.x;
  if (d == 0) {
    float mx = -3.0e38f;
    for (int k = 0; k < S_; ++k) mx = fmaxf(mx, scores[g * S_ + k]);
    float s = 0.f;
    for (int k = 0; k < S_; ++k) { float e = __expf(scores[g * S_ + k] - mx); al[k] = e; s += e; }
    float inv = 1.f / s;
    for (int k = 0; k < S_; ++k) al[k] *= inv;
  }
  __syncthreads();
  const _Float16* nb = neigh + (size_t)g * S_ * D_;
  float a = 0.f;
  for (int k = 0; k < S_; ++k) a += al[k] * (float)nb[k * D_ + d];
  agg[(size_t)g * D_ + d] = (_Float16)a;
}

// out = relu([self | agg] @ w3); packed w3 (256x128) staged to LDS by TDM.
// 8 waves/block, 8 consecutive 16-row tiles (M divisible by 16, w3 shared).
__global__ void k_concat(const _Float16* __restrict__ self, const _Float16* __restrict__ agg,
                         const _Float16* __restrict__ pbW3, _Float16* __restrict__ out16,
                         float* __restrict__ outAdd, int mode, int totalTiles) {
  __shared__ __align__(32) _Float16 wlds[32768];   // 64 KB packed w3
  int tid = threadIdx.x;
  int lane = tid & 31, wave = tid >> 5;
  if (wave == 0) {
    tdm_load_to_lds(lds_offset_of(wlds), pbW3, 65536u);
    __builtin_amdgcn_s_wait_tensorcnt(0);
  }
  __syncthreads();
  int tile = blockIdx.x * 8 + wave;
  if (tile >= totalTiles) return;
  int r0 = tile * 16;
  int m = lane & 15, half = lane >> 4;
  const _Float16* As = self + (size_t)(r0 + m) * D_;
  const _Float16* Aa = agg  + (size_t)(r0 + m) * D_;
  v16h afr[8];
#pragma unroll
  for (int kt = 0; kt < 4; ++kt) afr[kt]     = load_a_frag(As, kt * 32, half);
#pragma unroll
  for (int kt = 0; kt < 4; ++kt) afr[4 + kt] = load_a_frag(Aa, kt * 32, half);
#pragma unroll
  for (int nt = 0; nt < 8; ++nt) {
    v8f c = vzero8();
#pragma unroll
    for (int kt = 0; kt < 8; ++kt) {
      v16h bf = *(const v16h*)(wlds + (((kt * 8 + nt) * 32 + lane) * 16));
      c = wmma32(afr[kt], bf, c);
    }
#pragma unroll
    for (int i = 0; i < 8; ++i) {
      int row = r0 + half * 8 + i, col = nt * 16 + m;
      float v = fmaxf(c[i], 0.f);
      if (mode == 0) out16[(size_t)row * D_ + col] = (_Float16)v;
      else           outAdd[(size_t)row * D_ + col] += v;   // h_local + h_global
    }
  }
}

// ---------------- host launcher ----------------
extern "C" void kernel_launch(void* const* d_in, const int* in_sizes, int n_in,
                              void* d_out, int out_size, void* d_ws, size_t ws_size,
                              hipStream_t stream) {
  const int*   inputs    = (const int*)d_in[0];
  const int*   adj       = (const int*)d_in[1];
  const int*   mask      = (const int*)d_in[2];
  const int*   item      = (const int*)d_in[3];
  const int*   first_adj = (const int*)d_in[4];
  const int*   adj_all   = (const int*)d_in[5];
  const float* emb       = (const float*)d_in[6];
  const float* rel       = (const float*)d_in[7];
  const float* la        = (const float*)d_in[8];
  const float* w1        = (const float*)d_in[9];
  const float* w2        = (const float*)d_in[10];
  const float* w3        = (const float*)d_in[11];
  float* out = (float*)d_out;

  char* ws = (char*)d_ws;
  size_t off = 0;
  auto alloc = [&](size_t bytes) -> void* {
    void* p = ws + off;
    off = (off + bytes + 255) & ~(size_t)255;
    return p;
  };
  _Float16* h16   = (_Float16*)alloc((size_t)B_ * L_ * D_ * 2);
  _Float16* it16  = (_Float16*)alloc((size_t)B_ * L_ * D_ * 2);
  float*    it32  = (float*)   alloc((size_t)B_ * L_ * D_ * 4);
  float*    s_b   = (float*)   alloc((size_t)B_ * D_ * 4);
  _Float16* pbW1  = (_Float16*)alloc((size_t)2 * B_ * 16384 * 2);
  _Float16* pbW3  = (_Float16*)alloc((size_t)2 * 32768 * 2);
  _Float16* ev1   = (_Float16*)alloc((size_t)B_ * M1_ * D_ * 2);
  _Float16* ev2   = (_Float16*)alloc((size_t)B_ * M2_ * D_ * 2);
  float*    score = (float*)   alloc((size_t)B_ * M2_ * 4);
  _Float16* agg   = (_Float16*)alloc((size_t)B_ * M1_ * D_ * 2);
  _Float16* out00 = (_Float16*)alloc((size_t)B_ * L_ * D_ * 2);
  _Float16* out01 = (_Float16*)alloc((size_t)B_ * M1_ * D_ * 2);

  k_prep  <<<B_ * L_, 128, 0, stream>>>(inputs, item, mask, emb, h16, it16, it32);
  k_mean  <<<B_,      128, 0, stream>>>(it32, mask, s_b);
  k_packw1<<<2 * B_,  256, 0, stream>>>(w1, s_b, pbW1);
  k_packw3<<<2,       256, 0, stream>>>(w3, pbW3);

  k_local <<<B_, 256, 0, stream>>>(h16, adj, la, out);                     // h_local
  k_attn  <<<B_, 256, 0, stream>>>(it16, rel, out + (size_t)B_ * L_ * D_); // output_emb

  k_gather1<<<(B_ * M1_) / 4, 128, 0, stream>>>(first_adj, emb, ev1);
  k_gather2<<<(B_ * M2_) / 4, 128, 0, stream>>>(first_adj, adj_all, emb, ev2);

  const int TPB1 = (M1_ + 15) / 16, BPB1 = (TPB1 + 7) / 8;   // 38, 5
  const int TPB2 = (M2_ + 15) / 16, BPB2 = (TPB2 + 7) / 8;   // 450, 57
  const int T00 = (B_ * L_) / 16;    // 400 concat tiles
  const int T01 = (B_ * M1_) / 16;   // 4800 concat tiles

  // n_hop = 0, hop = 0: self = h, neigh = ev1
  k_score  <<<B_ * BPB1, 256, 0, stream>>>(ev1, pbW1, w2, score, M1_, TPB1, BPB1);
  k_softagg<<<B_ * L_, 128, 0, stream>>>(score, ev1, agg);
  k_concat <<<(T00 + 7) / 8, 256, 0, stream>>>(h16, agg, pbW3, out00, nullptr, 0, T00);

  // n_hop = 0, hop = 1: self = ev1, neigh = ev2
  k_score  <<<B_ * BPB2, 256, 0, stream>>>(ev2, pbW1, w2, score, M2_, TPB2, BPB2);
  k_softagg<<<B_ * M1_, 128, 0, stream>>>(score, ev2, agg);
  k_concat <<<(T01 + 7) / 8, 256, 0, stream>>>(ev1, agg, pbW3, out01, nullptr, 0, T01);

  // n_hop = 1, hop = 0: self = out00, neigh = out01; add h_global into out[0]
  k_score  <<<B_ * BPB1, 256, 0, stream>>>(out01, pbW1 + (size_t)B_ * 16384,
                                           w2 + D_, score, M1_, TPB1, BPB1);
  k_softagg<<<B_ * L_, 128, 0, stream>>>(score, out01, agg);
  k_concat <<<(T00 + 7) / 8, 256, 0, stream>>>(out00, agg, pbW3 + 32768, nullptr, out, 1, T00);
}
